// ParticleNet_73358041416059
// MI455X (gfx1250) — matrix operand order repaired and checked by reading, not compiled, not run on hardware
//
#include <hip/hip_runtime.h>
#include <hip/hip_bf16.h>

typedef __attribute__((ext_vector_type(16))) __bf16 v16bf;
typedef __attribute__((ext_vector_type(8)))  float  v8f;
typedef __attribute__((ext_vector_type(2)))  float  v2f;

#define BN_EPS 1e-5f

// ============================================================
// Weight prep: fold BatchNorm into Linear, cast to bf16, and
// pre-swizzle into the WMMA B-fragment per-lane layout:
//   elem index = ((nt*numKt + kt)*32 + lane)*16 + e
//   with  k = kt*32 + (lane/16)*16 + e,  n = nt*16 + (lane%16)
// so each lane's 16 bf16 B-fragment values are 32 contiguous bytes.
// ============================================================
__global__ __launch_bounds__(256)
void prep_layer_kernel(const float* __restrict__ W,
                       const float* __restrict__ b,
                       const float* __restrict__ beta,
                       const float* __restrict__ gamma,
                       const float* __restrict__ mean,
                       const float* __restrict__ var,
                       __bf16* __restrict__ wswz,
                       float* __restrict__ bout,
                       int din, int dout, int kpad)
{
    int tid   = blockIdx.x * blockDim.x + threadIdx.x;
    int total = kpad * dout;
    int numKt = kpad >> 5;

    if (tid < dout) {
        float s = gamma[tid] * rsqrtf(var[tid] + BN_EPS);
        bout[tid] = (b[tid] - mean[tid]) * s + beta[tid];
    }
    for (int e = tid; e < total; e += blockDim.x * gridDim.x) {
        int frag   = e >> 9;
        int within = e & 511;
        int lane   = within >> 4;
        int el     = within & 15;
        int nt     = frag / numKt;
        int kt     = frag - nt * numKt;
        int kin    = ((lane >> 4) << 4) + el;
        int n      = (nt << 4) + (lane & 15);
        int k      = (kt << 5) + kin;
        float v = 0.f;
        if (k < din) {
            float s = gamma[n] * rsqrtf(var[n] + BN_EPS);
            v = W[(size_t)k * dout + n] * s;
        }
        wswz[e] = (__bf16)v;
    }
}

// ============================================================
// kNN: Gram matrix P.P^T per batch via fp32 WMMA (16x16x4_f32),
// then d = |p|^2 + |q|^2 - 2 dot (the reference formula, full fp32
// so top-k ordering matches), then branchless register top-16.
// A(m,k): lane = m%16 + 16*(k/2), vgpr = k%2  -> contiguous float2
// B(k,n): lane = n%16 + 16*(k/2), vgpr = k%2  -> same load pattern
// ============================================================
template<int F>
__global__ __launch_bounds__(256)
void knn_kernel(const float* __restrict__ h, int D, int* __restrict__ idxOut)
{
    constexpr int Fp = (F + 3) & ~3;
    extern __shared__ char dynsmem[];
    float* pts   = (float*)dynsmem;           // 128 * Fp
    float* dist  = pts + 128 * Fp;            // 128 * 128 (dot products)
    float* norms = dist + 128 * 128;          // 128

    int b   = blockIdx.x;
    int tid = threadIdx.x;

    for (int e = tid; e < 128 * Fp; e += 256) {
        int p = e / Fp;
        int f = e - p * Fp;
        pts[e] = (f < F) ? h[((size_t)((b << 7) + p)) * D + f] : 0.f;
    }
    __syncthreads();

    if (tid < 128) {
        const float* r = pts + tid * Fp;
        float s = 0.f;
#pragma unroll 4
        for (int f = 0; f < Fp; ++f) s += r[f] * r[f];
        norms[tid] = s;
    }

    // Gram via f32 WMMA: 8x8 tiles of 16x16, K chunks of 4
    int wave  = tid >> 5;
    int lane  = tid & 31;
    int laneM = lane & 15;
    int hi    = lane >> 4;
    for (int t = wave; t < 64; t += 8) {
        int mt = t & 7;
        int nt = t >> 3;
        v8f acc = {};
        const float* arow = pts + (mt * 16 + laneM) * Fp + hi * 2;
        const float* brow = pts + (nt * 16 + laneM) * Fp + hi * 2;
#pragma unroll
        for (int kc = 0; kc < Fp / 4; ++kc) {
            v2f a  = *(const v2f*)(arow + kc * 4);
            v2f bb = *(const v2f*)(brow + kc * 4);
            acc = __builtin_amdgcn_wmma_f32_16x16x4_f32(
                      false, a, false, bb, (short)0, acc, false, false);
        }
        int mb = mt * 16 + (hi ? 8 : 0);
#pragma unroll
        for (int r = 0; r < 8; ++r)
            dist[(mb + r) * 128 + (nt << 4) + laneM] = acc[r];
    }
    __syncthreads();

    if (tid < 128) {
        int p = tid;
        float np = norms[p];
        const float* drow = dist + p * 128;
        float bd[16]; int bi[16];
#pragma unroll
        for (int j = 0; j < 16; ++j) { bd[j] = 3.0e38f; bi[j] = 0; }
        for (int q = 0; q < 128; ++q) {
            float d = np + norms[q] - 2.f * drow[q];
            d = (q == p) ? 3.0e38f : d;        // branchless self-exclusion
            if (d < bd[15]) {
                bd[15] = d; bi[15] = q;
#pragma unroll
                for (int j = 15; j > 0; --j) {
                    if (bd[j] < bd[j - 1]) {
                        float td = bd[j]; bd[j] = bd[j - 1]; bd[j - 1] = td;
                        int   ti = bi[j]; bi[j] = bi[j - 1]; bi[j - 1] = ti;
                    }
                }
            }
        }
#pragma unroll
        for (int j = 0; j < 16; ++j)
            idxOut[(((b << 7) + p) << 4) + j] = bi[j];
    }
}

// ============================================================
// LDS(A) x Global(B, pre-swizzled) bf16 WMMA GEMM, M=128.
// Each wave owns one 16-row M stripe: all K-chunk A fragments are
// hoisted into registers once, reused across every N tile.
// ============================================================
template<int KDIM, int NDIM>
__device__ __forceinline__
void gemm_lds(const __bf16* ldsA,
              const __bf16* __restrict__ wswz,
              const float* __restrict__ bias,
              __bf16* ldsOut, int tid)
{
    constexpr int numKt = KDIM >> 5;
    constexpr int numNt = NDIM >> 4;

    int wave   = tid >> 5;                 // = mt (8 waves, 8 M tiles)
    int lane   = tid & 31;
    int laneM  = lane & 15;
    int hiHalf = lane >> 4;
    int aKoff  = hiHalf ? 8 : 0;

    union Afrag { v16bf v; unsigned u[8]; };
    Afrag afr[numKt];
    const __bf16* arow = ldsA + (wave * 16 + laneM) * KDIM;
#pragma unroll
    for (int kt = 0; kt < numKt; ++kt) {
        const __bf16* ab = arow + (kt << 5);
#pragma unroll
        for (int v = 0; v < 8; ++v) {
            int kp = (aKoff << 1) + (v << 1);
            afr[kt].u[v] = *(const unsigned*)(const void*)(ab + kp);
        }
    }

    const __bf16* wlane = wswz + (lane << 4);
    for (int nt = 0; nt < numNt; ++nt) {
        v8f acc = {};
        const __bf16* wfrag = wlane + ((size_t)(nt * numKt) << 9);
#pragma unroll
        for (int kt = 0; kt < numKt; ++kt) {
            v16bf bfrag = *(const v16bf*)(const void*)(wfrag + ((size_t)kt << 9));
            __builtin_prefetch(wfrag + ((size_t)(kt + numKt) << 9), 0, 0);
            acc = __builtin_amdgcn_wmma_f32_16x16x32_bf16(
                      false, afr[kt].v, false, bfrag, (short)0, acc, false, false);
        }
        int   ncol = (nt << 4) + laneM;
        float bv   = bias[ncol];
        int   mb   = wave * 16 + (hiHalf ? 8 : 0);
#pragma unroll
        for (int r = 0; r < 8; ++r) {
            float vv = acc[r] + bv;
            vv = vv > 0.f ? vv : 0.f;
            ldsOut[(mb + r) * NDIM + ncol] = (__bf16)vv;
        }
    }
}

// ============================================================
// EdgeConv block: 1 workgroup = 8 points x 16 neighbors = 128 rows.
// [x_i, x_j - x_i] -> 3x (WMMA + bias + ReLU) -> mean over K
// -> write [edge_out, h_prev] concat.   DIN == ins, DOUT = OUT+DIN.
// ============================================================
template<int DIN, int OUT, int KPAD1>
__global__ __launch_bounds__(256)
void edgeconv_kernel(const float* __restrict__ hIn,
                     const int* __restrict__ idx,
                     const __bf16* __restrict__ w1,
                     const __bf16* __restrict__ w2,
                     const __bf16* __restrict__ w3,
                     const float* __restrict__ b1,
                     const float* __restrict__ b2,
                     const float* __restrict__ b3,
                     float* __restrict__ hOut)
{
    constexpr int DOUT = OUT + DIN;
    extern __shared__ char dynsmem[];
    __bf16* ldsA = (__bf16*)dynsmem;           // 128 * KPAD1
    __bf16* buf2 = ldsA + 128 * KPAD1;         // 128 * OUT
    __bf16* buf3 = buf2 + 128 * OUT;           // 128 * OUT

    int tid = threadIdx.x;
    int b   = blockIdx.x >> 4;
    int g   = blockIdx.x & 15;

    // ---- stage edge features (fp32 -> bf16, packed u32 stores) ----
    {
        int r    = tid & 127;
        int half = tid >> 7;
        int pi   = (g << 3) + (r >> 4);
        int kk   = r & 15;
        int gpi  = (b << 7) + pi;
        int j    = idx[(gpi << 4) + kk];
        const float* xi = hIn + (size_t)gpi * DIN;
        const float* xj = hIn + (size_t)((b << 7) + j) * DIN;
        constexpr int ck = KPAD1 >> 1;
        int c0 = half * ck;
        __bf16* dst = ldsA + r * KPAD1;
        auto ev = [&](int c) -> float {
            if (c < DIN)          return xi[c];
            else if (c < 2 * DIN) return xj[c - DIN] - xi[c - DIN];
            return 0.f;
        };
#pragma unroll 4
        for (int c = c0; c < c0 + ck; c += 2) {
            union { __bf16 hh[2]; unsigned u; } pk;
            pk.hh[0] = (__bf16)ev(c);
            pk.hh[1] = (__bf16)ev(c + 1);
            *(unsigned*)(void*)(dst + c) = pk.u;
        }
    }
    __syncthreads();

    gemm_lds<KPAD1, OUT>(ldsA, w1, b1, buf2, tid);
    __syncthreads();
    gemm_lds<OUT, OUT>(buf2, w2, b2, buf3, tid);
    __syncthreads();
    gemm_lds<OUT, OUT>(buf3, w3, b3, buf2, tid);
    __syncthreads();

    // ---- mean over K=16 messages, write + concat previous features ----
    for (int e = tid; e < 8 * OUT; e += 256) {
        int pl = e / OUT;
        int n  = e - pl * OUT;
        const __bf16* rows = buf2 + ((pl << 4) * OUT) + n;
        float s = 0.f;
#pragma unroll
        for (int r = 0; r < 16; ++r) s += (float)rows[r * OUT];
        hOut[(size_t)((b << 7) + (g << 3) + pl) * DOUT + n] = s * (1.f / 16.f);
    }
    for (int e = tid; e < 8 * DIN; e += 256) {
        int pl = e / DIN;
        int c  = e - pl * DIN;
        int gp = (b << 7) + (g << 3) + pl;
        hOut[(size_t)gp * DOUT + OUT + c] = hIn[(size_t)gp * DIN + c];
    }
}

// ============================================================
// Global mean pool over particles + fc1(ReLU) + fc2. Tiny: VALU.
// ============================================================
__global__ __launch_bounds__(256)
void pool_fc_kernel(const float* __restrict__ h,
                    const float* __restrict__ fc1W, const float* __restrict__ fc1b,
                    const float* __restrict__ fc2W, const float* __restrict__ fc2b,
                    float* __restrict__ outp)
{
    __shared__ float gbuf[455];
    __shared__ float a1[256];
    int b   = blockIdx.x;
    int tid = threadIdx.x;
    const float* hb = h + (size_t)b * 128 * 455;
    for (int d = tid; d < 455; d += 256) {
        float s = 0.f;
        for (int p = 0; p < 128; ++p) s += hb[(size_t)p * 455 + d];
        gbuf[d] = s * (1.f / 128.f);
    }
    __syncthreads();
    {
        float s = fc1b[tid];
        for (int d = 0; d < 455; ++d) s += gbuf[d] * fc1W[(size_t)d * 256 + tid];
        a1[tid] = s > 0.f ? s : 0.f;
    }
    __syncthreads();
    if (tid < 5) {
        float s = fc2b[tid];
        for (int d = 0; d < 256; ++d) s += a1[d] * fc2W[(size_t)d * 5 + tid];
        outp[b * 5 + tid] = s;
    }
}

// ============================================================
// Host orchestration
// ============================================================
extern "C" void kernel_launch(void* const* d_in, const int* in_sizes, int n_in,
                              void* d_out, int out_size, void* d_ws, size_t ws_size,
                              hipStream_t stream)
{
    (void)in_sizes; (void)n_in; (void)out_size; (void)ws_size;

    const int B = 64, P = 128;
    const int outsA[3] = {64, 128, 256};
    const int din1 [3] = {14, 142, 398};
    const int kpadA[3] = {32, 160, 416};
    const int Dh   [4] = {7, 71, 199, 455};

    const float* x = (const float*)d_in[0];
    #define LP(i, l, f) ((const float*)d_in[1 + ((i) * 3 + (l)) * 6 + (f)])
    const float* fc1W = (const float*)d_in[55];
    const float* fc1b = (const float*)d_in[56];
    const float* fc2W = (const float*)d_in[57];
    const float* fc2b = (const float*)d_in[58];

    char*  base = (char*)d_ws;
    size_t off  = 0;
    auto alloc = [&](size_t bytes) -> char* {
        off = (off + 255) & ~(size_t)255;
        char* p = base + off;
        off += bytes;
        return p;
    };

    __bf16* wswz[3][3];
    float*  bfold[3][3];
    for (int i = 0; i < 3; ++i)
        for (int l = 0; l < 3; ++l) {
            int kpad = (l == 0) ? kpadA[i] : outsA[i];
            wswz[i][l]  = (__bf16*)alloc((size_t)kpad * outsA[i] * sizeof(__bf16));
            bfold[i][l] = (float*)alloc((size_t)outsA[i] * sizeof(float));
        }
    int*   idxbuf = (int*)alloc((size_t)B * P * 16 * sizeof(int));
    float* h1 = (float*)alloc((size_t)B * P * Dh[1] * sizeof(float));
    float* h2 = (float*)alloc((size_t)B * P * Dh[2] * sizeof(float));
    float* h3 = (float*)alloc((size_t)B * P * Dh[3] * sizeof(float));

    for (int i = 0; i < 3; ++i)
        for (int l = 0; l < 3; ++l) {
            int din  = (l == 0) ? din1[i]  : outsA[i];
            int kpad = (l == 0) ? kpadA[i] : outsA[i];
            int dout = outsA[i];
            int total = kpad * dout;
            int grid  = (total + 255) / 256;
            prep_layer_kernel<<<grid, 256, 0, stream>>>(
                LP(i, l, 0), LP(i, l, 1), LP(i, l, 2),
                LP(i, l, 3), LP(i, l, 4), LP(i, l, 5),
                wswz[i][l], bfold[i][l], din, dout, kpad);
        }

    auto knn_smem = [](int Fp) { return (size_t)(128 * Fp + 128 * 128 + 128) * sizeof(float); };
    auto ec_smem  = [](int kpad, int out) { return (size_t)(128 * kpad + 2 * 128 * out) * sizeof(__bf16); };

    // ---- block 0 ----
    knn_kernel<2><<<B, 256, knn_smem(4), stream>>>(x, 7, idxbuf);
    edgeconv_kernel<7, 64, 32><<<B * 16, 256, ec_smem(32, 64), stream>>>(
        x, idxbuf, wswz[0][0], wswz[0][1], wswz[0][2],
        bfold[0][0], bfold[0][1], bfold[0][2], h1);

    // ---- block 1 ----
    knn_kernel<71><<<B, 256, knn_smem(72), stream>>>(h1, 71, idxbuf);
    edgeconv_kernel<71, 128, 160><<<B * 16, 256, ec_smem(160, 128), stream>>>(
        h1, idxbuf, wswz[1][0], wswz[1][1], wswz[1][2],
        bfold[1][0], bfold[1][1], bfold[1][2], h2);

    // ---- block 2 ----
    knn_kernel<199><<<B, 256, knn_smem(200), stream>>>(h2, 199, idxbuf);
    edgeconv_kernel<199, 256, 416><<<B * 16, 256, ec_smem(416, 256), stream>>>(
        h2, idxbuf, wswz[2][0], wswz[2][1], wswz[2][2],
        bfold[2][0], bfold[2][1], bfold[2][2], h3);

    pool_fc_kernel<<<B, 256, 0, stream>>>(h3, fc1W, fc1b, fc2W, fc2b, (float*)d_out);
    #undef LP
}
